// Decoder_21105469293006
// MI455X (gfx1250) — compile-verified, matrix operand order checked by reading, not run on hardware
//
#include <hip/hip_runtime.h>
#include <hip/hip_bf16.h>
#include <math.h>

// ---------------------------------------------------------------------------
// Types for CDNA5 WMMA (wave32): A/B = 16 x bf16 (8 VGPRs), C/D = 8 x f32
// ---------------------------------------------------------------------------
typedef __attribute__((ext_vector_type(16))) __bf16 v16bf;
typedef __attribute__((ext_vector_type(8)))  float  v8f;

#define BM  128
#define BN  128
#define BK  32
#define BKP 40   // padded LDS K-stride (80 B, multiple of 16 B)

// Model constants
#define Bsz 8
#define Ssz 256
#define Tsz 256
#define Dsz 512
#define Hsz 8
#define Vsz 32000
#define Dff 2048
#define VOUT 32001

struct GemmArgs {
  const float* A;      // M x K row-major (f32, converted to bf16 in staging)
  const float* Bm;     // opB==0: K x N row-major ; opB==1: N x K row-major
  const float* bias;   // length N (may be null)
  float*       C;      // M x N (ldc)
  int M, N, K, lda, ldb, ldc;
  long long sA1, sA2, sB1, sB2, sBias1, sBias2, sC1, sC2; // batch strides
  int binner;          // inner batch count (z = z1*binner + z2)
  int opB;
  int epi;             // 0 none, 1 +bias, 2 +bias+relu
};

// Pack a float4 into 4 bf16 (one 8-byte LDS store)
union Pk4 { __bf16 h[4]; uint2 u; };
__device__ __forceinline__ uint2 pack_bf16x4(float4 f) {
  Pk4 p;
  p.h[0] = (__bf16)f.x; p.h[1] = (__bf16)f.y;
  p.h[2] = (__bf16)f.z; p.h[3] = (__bf16)f.w;
  return p.u;
}

// ---------------------------------------------------------------------------
// Batched GEMM: C = A * B (+bias)(+relu), bf16 WMMA with f32 accumulation.
// Workgroup: 256 threads = 8 waves; WG tile 128x128, K-step 32.
// Wave grid 4(M) x 2(N); each wave: 32x64 via 2x4 v_wmma_f32_16x16x32_bf16.
// ---------------------------------------------------------------------------
__global__ __launch_bounds__(256) void gemm_bf16_wmma(GemmArgs g) {
  __shared__ __align__(16) __bf16 As[BM][BKP];
  __shared__ __align__(16) __bf16 Bs[BN][BKP];

  const int tid = threadIdx.x;
  const int n0  = blockIdx.x * BN;
  const int m0  = blockIdx.y * BM;
  const int z   = blockIdx.z;
  const int z1  = z / g.binner;
  const int z2  = z % g.binner;

  const float* Ap = g.A  + z1 * g.sA1 + z2 * g.sA2;
  const float* Bp = g.Bm + z1 * g.sB1 + z2 * g.sB2;
  float*       Cp = g.C  + z1 * g.sC1 + z2 * g.sC2;
  const float* biasp = g.bias ? (g.bias + z1 * g.sBias1 + z2 * g.sBias2) : nullptr;

  const int lane = tid & 31;
  const int wid  = tid >> 5;
  const int wm   = (wid & 3) * 32;   // 4 waves along M
  const int wn   = (wid >> 2) * 64;  // 2 waves along N

  v8f acc[2][4];
  #pragma unroll
  for (int i = 0; i < 2; ++i)
    #pragma unroll
    for (int j = 0; j < 4; ++j)
      acc[i][j] = (v8f){0.f, 0.f, 0.f, 0.f, 0.f, 0.f, 0.f, 0.f};

  const int lr = lane & 15;          // fragment row / col within 16
  const int kb = (lane >> 4) << 3;   // 0 or 8: K-half per 16-bit WMMA layout

  for (int k0 = 0; k0 < g.K; k0 += BK) {
    // ---- stage A tile (BM x BK), f32 -> bf16, packed b64 LDS stores ----
    #pragma unroll
    for (int i = 0; i < 4; ++i) {
      int idx = tid + i * 256;        // 0..1023 (8 float4 per row)
      int row = idx >> 3;
      int c4  = (idx & 7) << 2;
      float4 f = make_float4(0.f, 0.f, 0.f, 0.f);
      if (m0 + row < g.M) {
        const float* p = Ap + (long long)(m0 + row) * g.lda + k0 + c4;
        if (k0 + BK < g.K) __builtin_prefetch(p + BK, 0, 1); // global_prefetch_b8
        f = *(const float4*)p;
      }
      *(uint2*)&As[row][c4] = pack_bf16x4(f);
    }
    // ---- stage B tile into Bs[n][k] ----
    if (g.opB) {
      // Bm is N x K row-major: straight rows, coalesced along K
      #pragma unroll
      for (int i = 0; i < 4; ++i) {
        int idx = tid + i * 256;
        int row = idx >> 3;
        int c4  = (idx & 7) << 2;
        float4 f = make_float4(0.f, 0.f, 0.f, 0.f);
        if (n0 + row < g.N)
          f = *(const float4*)(Bp + (long long)(n0 + row) * g.ldb + k0 + c4);
        *(uint2*)&Bs[row][c4] = pack_bf16x4(f);
      }
    } else {
      // Bm is K x N row-major: float4 along N (coalesced), transpose-scatter
      #pragma unroll
      for (int i = 0; i < 4; ++i) {
        int idx = tid + i * 256;      // 0..1023
        int kk  = idx >> 5;           // 0..31
        int n4  = (idx & 31) << 2;    // 0,4,...,124
        const float* p = Bp + (long long)(k0 + kk) * g.ldb + n0 + n4;
        float4 f = make_float4(0.f, 0.f, 0.f, 0.f);
        if (n0 + n4 + 3 < g.N) {
          f = *(const float4*)p;
        } else {
          if (n0 + n4 + 0 < g.N) f.x = p[0];
          if (n0 + n4 + 1 < g.N) f.y = p[1];
          if (n0 + n4 + 2 < g.N) f.z = p[2];
          if (n0 + n4 + 3 < g.N) f.w = p[3];
        }
        Bs[n4 + 0][kk] = (__bf16)f.x;
        Bs[n4 + 1][kk] = (__bf16)f.y;
        Bs[n4 + 2][kk] = (__bf16)f.z;
        Bs[n4 + 3][kk] = (__bf16)f.w;
      }
    }
    __syncthreads();

    // ---- fragment loads (16-bit A layout: elems 0..7 = K kb..kb+7,
    //      elems 8..15 = K kb+16..kb+23) ----
    v16bf afrag[2], bfrag[4];
    #pragma unroll
    for (int fm = 0; fm < 2; ++fm) {
      const __bf16* p = &As[wm + fm * 16 + lr][kb];
      *((uint4*)&afrag[fm])     = *(const uint4*)p;
      *((uint4*)&afrag[fm] + 1) = *(const uint4*)(p + 16);
    }
    #pragma unroll
    for (int fn = 0; fn < 4; ++fn) {
      const __bf16* p = &Bs[wn + fn * 16 + lr][kb];
      *((uint4*)&bfrag[fn])     = *(const uint4*)p;
      *((uint4*)&bfrag[fn] + 1) = *(const uint4*)(p + 16);
    }
    #pragma unroll
    for (int fm = 0; fm < 2; ++fm)
      #pragma unroll
      for (int fn = 0; fn < 4; ++fn)
        acc[fm][fn] = __builtin_amdgcn_wmma_f32_16x16x32_bf16(
            false, afrag[fm], false, bfrag[fn], (short)0, acc[fm][fn],
            false, false);
    __syncthreads();
  }

  // ---- epilogue: C/D layout VGPR r -> M = r + 8*(lane>=16), N = lane%16 ----
  const int rbase = (lane >> 4) << 3;
  #pragma unroll
  for (int fn = 0; fn < 4; ++fn) {
    int col = n0 + wn + fn * 16 + lr;
    if (col >= g.N) continue;
    float bv = (g.epi >= 1 && biasp) ? biasp[col] : 0.f;
    #pragma unroll
    for (int fm = 0; fm < 2; ++fm) {
      #pragma unroll
      for (int r = 0; r < 8; ++r) {
        int row = m0 + wm + fm * 16 + rbase + r;
        if (row < g.M) {
          float vv = acc[fm][fn][r] + bv;
          if (g.epi == 2) vv = fmaxf(vv, 0.f);
          Cp[(long long)row * g.ldc + col] = vv;
        }
      }
    }
  }
}

// ---------------------------------------------------------------------------
// x[b,t,:] = emb[target[b,t],:] + sinusoidal_pos_enc(t,:)
// ---------------------------------------------------------------------------
__global__ __launch_bounds__(256) void embed_kernel(const int* __restrict__ tgt,
                                                    const float* __restrict__ emb,
                                                    float* __restrict__ x) {
  int i = blockIdx.x * 256 + threadIdx.x;       // B*T*D = 1,048,576
  int d  = i & (Dsz - 1);
  int bt = i >> 9;
  int t  = bt & (Tsz - 1);
  int tok = tgt[bt];
  float den = __expf(-(float)(d & ~1) * (9.210340371976184f / (float)Dsz));
  float ang = (float)t * den;
  float pe  = (d & 1) ? __cosf(ang) : __sinf(ang);
  x[i] = emb[(long long)tok * Dsz + d] + pe;
}

// ---------------------------------------------------------------------------
// out = LayerNorm(x + r) * g + b     (one 256-thread block per 512-wide row)
// ---------------------------------------------------------------------------
__global__ __launch_bounds__(256) void add_ln_kernel(const float* __restrict__ x,
                                                     const float* __restrict__ r,
                                                     const float* __restrict__ g,
                                                     const float* __restrict__ b,
                                                     float* __restrict__ out) {
  const int row = blockIdx.x;
  const int t   = threadIdx.x;
  __shared__ float red[256];
  const long long base = (long long)row * Dsz;
  float v0 = x[base + t]        + (r ? r[base + t]        : 0.f);
  float v1 = x[base + t + 256]  + (r ? r[base + t + 256]  : 0.f);
  red[t] = v0 + v1;
  __syncthreads();
  for (int o = 128; o > 0; o >>= 1) { if (t < o) red[t] += red[t + o]; __syncthreads(); }
  float mean = red[0] * (1.f / (float)Dsz);
  __syncthreads();
  float d0 = v0 - mean, d1 = v1 - mean;
  red[t] = d0 * d0 + d1 * d1;
  __syncthreads();
  for (int o = 128; o > 0; o >>= 1) { if (t < o) red[t] += red[t + o]; __syncthreads(); }
  float inv = rsqrtf(red[0] * (1.f / (float)Dsz) + 1e-5f);
  out[base + t]       = d0 * inv * g[t]       + b[t];
  out[base + t + 256] = d1 * inv * g[t + 256] + b[t + 256];
}

// ---------------------------------------------------------------------------
// In-place masked scaled softmax over rows of scores laid out [H,B,T,S].
// mode 0: causal * (tgt!=V)q * (tgt!=V)k ; mode 1: (tgt!=V)q * src_pad[k]
// ---------------------------------------------------------------------------
__global__ __launch_bounds__(256) void softmax_mask_kernel(float* __restrict__ sc,
                                                           const int* __restrict__ tgt,
                                                           const float* __restrict__ srcpad,
                                                           int mode, float scale) {
  const int r   = blockIdx.x;                 // H*B*T rows
  const int rem = r % (Bsz * Tsz);
  const int b   = rem / Tsz;
  const int q   = rem % Tsz;
  const int k   = threadIdx.x;                // S == 256
  __shared__ float red[256];
  float* row = sc + (long long)r * Ssz;

  bool tpq = (tgt[b * Tsz + q] != Vsz);
  bool valid;
  if (mode == 0) valid = tpq && (k <= q) && (tgt[b * Tsz + k] != Vsz);
  else           valid = tpq && (srcpad[b * Ssz + k] != 0.f);
  float s = valid ? row[k] * scale : -1e32f;

  red[k] = s;
  __syncthreads();
  for (int o = 128; o > 0; o >>= 1) { if (k < o) red[k] = fmaxf(red[k], red[k + o]); __syncthreads(); }
  float mx = red[0];
  __syncthreads();
  float e = __expf(s - mx);
  red[k] = e;
  __syncthreads();
  for (int o = 128; o > 0; o >>= 1) { if (k < o) red[k] += red[k + o]; __syncthreads(); }
  row[k] = e / red[0];
}

// probs [H,B,T,S]  ->  out [B,H,T,S]   (first block's cross-attn weights)
__global__ __launch_bounds__(256) void att_copy_kernel(const float* __restrict__ probs,
                                                       float* __restrict__ out) {
  long long i = (long long)blockIdx.x * 256 + threadIdx.x;  // H*B*T*S
  int k = (int)(i & 255);
  long long r = i >> 8;
  int q = (int)(r & 255);
  long long hb = r >> 8;
  int b = (int)(hb & 7);
  int h = (int)(hb >> 3);
  out[(((long long)b * Hsz + h) * Tsz + q) * Ssz + k] = probs[i];
}

// ---------------------------------------------------------------------------
// Host side
// ---------------------------------------------------------------------------
struct MhaP { const float *Wq,*bq,*Wk,*bk,*Wv,*bv,*Wo,*bo; };
struct BlkP { MhaP a1, a2; const float *g1,*b1,*g2,*b2,*g3,*b3,*W1,*bias1,*W2,*bias2; };

static void gemm(hipStream_t st, const float* A, const float* Bm, const float* bias,
                 float* C, int M, int N, int K, int lda, int ldb, int ldc,
                 long long sA1, long long sA2, long long sB1, long long sB2,
                 long long sBias1, long long sBias2, long long sC1, long long sC2,
                 int batch, int binner, int opB, int epi) {
  GemmArgs g;
  g.A = A; g.Bm = Bm; g.bias = bias; g.C = C;
  g.M = M; g.N = N; g.K = K; g.lda = lda; g.ldb = ldb; g.ldc = ldc;
  g.sA1 = sA1; g.sA2 = sA2; g.sB1 = sB1; g.sB2 = sB2;
  g.sBias1 = sBias1; g.sBias2 = sBias2; g.sC1 = sC1; g.sC2 = sC2;
  g.binner = binner; g.opB = opB; g.epi = epi;
  dim3 grid((N + BN - 1) / BN, (M + BM - 1) / BM, batch);
  gemm_bf16_wmma<<<grid, dim3(256), 0, st>>>(g);
}

extern "C" void kernel_launch(void* const* d_in, const int* in_sizes, int n_in,
                              void* d_out, int out_size, void* d_ws, size_t ws_size,
                              hipStream_t stream) {
  (void)in_sizes; (void)n_in; (void)out_size; (void)ws_size;

  const float* ES  = (const float*)d_in[0];   // encoded_source [B,S,D]
  const float* SP  = (const float*)d_in[1];   // source_padding [B,S]
  const int*   TGT = (const int*)  d_in[2];   // target [B,T]

  int pi = 3;
  auto nx = [&]() { return (const float*)d_in[pi++]; };
  const float* emb = nx();                    // [V+1, D]
  BlkP blk[4];
  for (int i = 0; i < 4; ++i) {
    MhaP* ms[2] = { &blk[i].a1, &blk[i].a2 };
    for (int m = 0; m < 2; ++m) {
      ms[m]->Wq = nx(); ms[m]->bq = nx();
      ms[m]->Wk = nx(); ms[m]->bk = nx();
      ms[m]->Wv = nx(); ms[m]->bv = nx();
      ms[m]->Wo = nx(); ms[m]->bo = nx();
    }
    blk[i].g1 = nx(); blk[i].b1 = nx();
    blk[i].g2 = nx(); blk[i].b2 = nx();
    blk[i].g3 = nx(); blk[i].b3 = nx();
    blk[i].W1 = nx(); blk[i].bias1 = nx();
    blk[i].W2 = nx(); blk[i].bias2 = nx();
  }
  const float* Wout = nx();
  const float* bout = nx();

  // Workspace layout (all f32)
  const long long BT   = (long long)Bsz * Tsz;       // 2048
  const long long BTD  = BT * Dsz;                   // 1,048,576
  const long long HBTD = (long long)Hsz * BTD;       // 8,388,608
  const long long HBTS = (long long)Hsz * Bsz * Tsz * Ssz; // 4,194,304
  float* X   = (float*)d_ws;        // [B*T, D]
  float* Q   = X   + BTD;           // [H, B*T, D]
  float* Kb  = Q   + HBTD;          // [H, B*T(or B*S), D]
  float* Vb  = Kb  + HBTD;
  float* SC  = Vb  + HBTD;          // [H, B, T, S]
  float* AT  = SC  + HBTS;          // [B*T, H*D] concat
  float* TMP = AT  + HBTD;          // [B*T, D]
  float* HID = Q;                   // reuse for FFN hidden [B*T, 4D]

  float* Y   = (float*)d_out;                       // [B*T, V+1]
  float* ATT = Y + BT * (long long)VOUT;            // [B,H,T,S]

  const float scale = 0.044194173824159216f;        // 1/sqrt(512)
  const long long DD = (long long)Dsz * Dsz;        // per-head weight stride
  const long long TD = (long long)Tsz * Dsz;        // 131072
  const long long TS = (long long)Tsz * Ssz;        // 65536

  // Embedding + positional encoding
  embed_kernel<<<(int)(BTD / 256), 256, 0, stream>>>(TGT, emb, X);

  for (int L = 0; L < 4; ++L) {
    const BlkP& P = blk[L];

    // ---------------- self attention ----------------
    gemm(stream, X, P.a1.Wq, P.a1.bq, Q,  2048, 512, 512, 512, 512, 512,
         0, 0, DD, 0, Dsz, 0, BTD, 0, Hsz, 1, 0, 1);
    gemm(stream, X, P.a1.Wk, P.a1.bk, Kb, 2048, 512, 512, 512, 512, 512,
         0, 0, DD, 0, Dsz, 0, BTD, 0, Hsz, 1, 0, 1);
    gemm(stream, X, P.a1.Wv, P.a1.bv, Vb, 2048, 512, 512, 512, 512, 512,
         0, 0, DD, 0, Dsz, 0, BTD, 0, Hsz, 1, 0, 1);
    // scores[h,b] = Q[h,b] (T x D) * K[h,b]^T  -> [H,B,T,S]
    gemm(stream, Q, Kb, nullptr, SC, 256, 256, 512, 512, 512, 256,
         BTD, TD, BTD, TD, 0, 0, (long long)Bsz * TS, TS, Hsz * Bsz, Bsz, 1, 0);
    softmax_mask_kernel<<<Hsz * Bsz * Tsz, 256, 0, stream>>>(SC, TGT, SP, 0, scale);
    // o[h,b] = probs (T x S) * V[h,b] (S x D), written into concat layout
    gemm(stream, SC, Vb, nullptr, AT, 256, 512, 256, 256, 512, Hsz * Dsz,
         (long long)Bsz * TS, TS, BTD, TD, 0, 0, Dsz, (long long)Tsz * Hsz * Dsz,
         Hsz * Bsz, Bsz, 0, 0);
    gemm(stream, AT, P.a1.Wo, P.a1.bo, TMP, 2048, 512, 4096, 4096, 512, 512,
         0, 0, 0, 0, 0, 0, 0, 0, 1, 1, 0, 1);
    add_ln_kernel<<<2048, 256, 0, stream>>>(X, TMP, P.g1, P.b1, X);

    // ---------------- cross attention ----------------
    gemm(stream, X,  P.a2.Wq, P.a2.bq, Q,  2048, 512, 512, 512, 512, 512,
         0, 0, DD, 0, Dsz, 0, BTD, 0, Hsz, 1, 0, 1);
    gemm(stream, ES, P.a2.Wk, P.a2.bk, Kb, 2048, 512, 512, 512, 512, 512,
         0, 0, DD, 0, Dsz, 0, BTD, 0, Hsz, 1, 0, 1);
    gemm(stream, ES, P.a2.Wv, P.a2.bv, Vb, 2048, 512, 512, 512, 512, 512,
         0, 0, DD, 0, Dsz, 0, BTD, 0, Hsz, 1, 0, 1);
    gemm(stream, Q, Kb, nullptr, SC, 256, 256, 512, 512, 512, 256,
         BTD, TD, BTD, TD, 0, 0, (long long)Bsz * TS, TS, Hsz * Bsz, Bsz, 1, 0);
    softmax_mask_kernel<<<Hsz * Bsz * Tsz, 256, 0, stream>>>(SC, TGT, SP, 1, scale);
    if (L == 0)
      att_copy_kernel<<<(int)(HBTS / 256), 256, 0, stream>>>(SC, ATT);
    gemm(stream, SC, Vb, nullptr, AT, 256, 512, 256, 256, 512, Hsz * Dsz,
         (long long)Bsz * TS, TS, BTD, TD, 0, 0, Dsz, (long long)Tsz * Hsz * Dsz,
         Hsz * Bsz, Bsz, 0, 0);
    gemm(stream, AT, P.a2.Wo, P.a2.bo, TMP, 2048, 512, 4096, 4096, 512, 512,
         0, 0, 0, 0, 0, 0, 0, 0, 1, 1, 0, 1);
    add_ln_kernel<<<2048, 256, 0, stream>>>(X, TMP, P.g2, P.b2, X);

    // ---------------- feed forward ----------------
    gemm(stream, X, P.W1, P.bias1, HID, 2048, 2048, 512, 512, 2048, 2048,
         0, 0, 0, 0, 0, 0, 0, 0, 1, 1, 0, 2);                       // bias+relu
    gemm(stream, HID, P.W2, P.bias2, TMP, 2048, 512, 2048, 2048, 512, 512,
         0, 0, 0, 0, 0, 0, 0, 0, 1, 1, 0, 1);
    add_ln_kernel<<<2048, 256, 0, stream>>>(X, TMP, P.g3, P.b3, X);
  }

  // Output projection: [2048,512] x [512,32001] + bout -> Y
  gemm(stream, X, Wout, bout, Y, 2048, VOUT, 512, 512, VOUT, VOUT,
       0, 0, 0, 0, 0, 0, 0, 0, 1, 1, 0, 1);
}
